// SSHConv3D_63591285785095
// MI455X (gfx1250) — matrix operand order repaired and checked by reading, not compiled, not run on hardware
//
#include <hip/hip_runtime.h>
#include <math.h>
#include <stdint.h>

typedef __attribute__((ext_vector_type(16))) _Float16 v16h;
typedef __attribute__((ext_vector_type(8)))  _Float16 v8h;
typedef __attribute__((ext_vector_type(8)))  float    v8f;

#define THREADS 128
#define NATOM   96     // 48 real + 48 imag atom columns
#define CIN     4
#define NF      16
#define DIMS    48

// workspace layout (bytes) -- atoms/wB/wpB are CONTIGUOUS (45,056 B total)
#define WS_ATOMS 0          // 12288 halfs: [kstep4][n96][k32]
#define WS_WB    24576      // 8192 halfs : [kh16][f16][k32]   (stage-2 B)
#define WS_WPB   40960      // 2048 halfs : [ks2][f16][k32]    (projection B)
#define WS_BANK_BYTES 45056

// LDS layout (bytes), total 125,824
#define L_XT    0           // 2944 halfs: 8^3 halo x 4ch + 896-half zero pad
#define L_AT    5888        // 24576 B atoms (async-copy dest, contiguous w/ next)
#define L_WB    30464       // 16384 B wB
#define L_WPB   46848       // 4096  B wpB
#define L_BIAS  50944
#define L_WCP   51200
#define L_BCP   51456
#define L_BPJ   51520
#define L_OFF   51584
#define L_RES2  52096       // 32768 halfs: [vox][im][kh][cr16]
#define L_SBUF  117632      // 4096 halfs : [vox][64]
#define L_TOTAL 125824

// ---------------------------------------------------------------------------
// Setup kernel 1: SH atom bank in f16, [kstep][n][k], taps 125..127 zeroed.
// ---------------------------------------------------------------------------
__global__ void sshconv_setup_atoms(_Float16* __restrict__ atoms) {
  int t = threadIdx.x;                 // tap index 0..127
  int a = t / 25; int rem = t - a * 25;
  int b = rem / 5; int c = rem - b * 5;
  // meshgrid(indexing='xy'): X = g[b], Y = g[a], Z = g[c]
  float px = (float)(b - 2), py = (float)(a - 2), pz = (float)(c - 2);
  float r  = sqrtf(px * px + py * py + pz * pz);
  float ct = (r > 0.f) ? (pz / r) : 0.f;
  float st = sqrtf(fmaxf(0.f, 1.f - ct * ct));
  float phi = atan2f(py, px);
  float c1p = cosf(phi), s1p = sinf(phi);
  float c2p = c1p * c1p - s1p * s1p, s2p = 2.f * c1p * s1p;
  float c3p = c2p * c1p - s2p * s1p, s3p = s2p * c1p + c2p * s1p;
  const float PI = 3.14159265358979f;
  float Yre[16], Yim[16];
  float st2 = st * st, st3 = st2 * st;
  float k00 = 0.5f   * sqrtf(1.f / PI);
  float k1  = 0.5f   * sqrtf(3.f / (2.f * PI));
  float k10 = 0.5f   * sqrtf(3.f / PI);
  float k22 = 0.25f  * sqrtf(15.f / (2.f * PI));
  float k21 = 0.5f   * sqrtf(15.f / (2.f * PI));
  float k20 = 0.25f  * sqrtf(5.f / PI);
  float k33 = 0.125f * sqrtf(35.f / PI);
  float k32 = 0.25f  * sqrtf(105.f / (2.f * PI));
  float k31 = 0.125f * sqrtf(21.f / PI);
  float k30 = 0.25f  * sqrtf(7.f / PI);
  float q   = 5.f * ct * ct - 1.f;
  Yre[0]  =  k00;                    Yim[0]  = 0.f;
  Yre[1]  =  k1 * st * c1p;          Yim[1]  = -k1 * st * s1p;
  Yre[2]  =  k10 * ct;               Yim[2]  = 0.f;
  Yre[3]  = -k1 * st * c1p;          Yim[3]  = -k1 * st * s1p;
  Yre[4]  =  k22 * st2 * c2p;        Yim[4]  = -k22 * st2 * s2p;
  Yre[5]  =  k21 * st * ct * c1p;    Yim[5]  = -k21 * st * ct * s1p;
  Yre[6]  =  k20 * (3.f * ct * ct - 1.f); Yim[6] = 0.f;
  Yre[7]  = -k21 * st * ct * c1p;    Yim[7]  = -k21 * st * ct * s1p;
  Yre[8]  =  k22 * st2 * c2p;        Yim[8]  =  k22 * st2 * s2p;
  Yre[9]  =  k33 * st3 * c3p;        Yim[9]  = -k33 * st3 * s3p;
  Yre[10] =  k32 * st2 * ct * c2p;   Yim[10] = -k32 * st2 * ct * s2p;
  Yre[11] =  k31 * st * q * c1p;     Yim[11] = -k31 * st * q * s1p;
  Yre[12] =  k30 * (5.f * ct * ct * ct - 3.f * ct); Yim[12] = 0.f;
  Yre[13] = -k31 * st * q * c1p;     Yim[13] = -k31 * st * q * s1p;
  Yre[14] =  k32 * st2 * ct * c2p;   Yim[14] =  k32 * st2 * ct * s2p;
  Yre[15] = -k33 * st3 * c3p;        Yim[15] = -k33 * st3 * s3p;
  float rad[3];
  rad[0] = fmaxf(0.f, 1.f - fabsf(r - 0.f));
  rad[1] = fmaxf(0.f, 1.f - fabsf(r - 1.f));
  rad[2] = fmaxf(0.f, 1.f - fabsf(r - 2.f));
  float valid = (t < 125) ? 1.f : 0.f;
  int kstep = t >> 5, k = t & 31;
  for (int rr = 0; rr < 3; rr++)
    for (int kh = 0; kh < 16; kh++) {
      int nre = rr * 16 + kh;          // column order: (im,r)*16 + kh
      int nim = 48 + rr * 16 + kh;
      atoms[(kstep * NATOM + nre) * 32 + k] = (_Float16)(valid * rad[rr] * Yre[kh]);
      atoms[(kstep * NATOM + nim) * 32 + k] = (_Float16)(valid * rad[rr] * Yim[kh]);
    }
}

// ---------------------------------------------------------------------------
// Setup kernel 2: runtime weights -> WMMA-B-layout f16 banks.
// ---------------------------------------------------------------------------
__global__ void sshconv_setup_weights(const float* __restrict__ w,
                                      const float* __restrict__ w_proj,
                                      _Float16* __restrict__ ws) {
  _Float16* wB  = ws + WS_WB  / 2;
  _Float16* wpB = ws + WS_WPB / 2;
  int tid = threadIdx.x;
  for (int e = tid; e < 8192; e += 256) {
    int k = e & 31, f = (e >> 5) & 15, kh = e >> 9;
    float v = 0.f;
    if (k < 12) {
      int c = k / 3, rr = k - c * 3;
      int deg = (kh == 0) ? 0 : ((kh < 4) ? 1 : ((kh < 9) ? 2 : 3));
      v = w[((c * NF + f) * 3 + rr) * 4 + deg];
    }
    wB[e] = (_Float16)v;
  }
  for (int e = tid; e < 1024; e += 256) {
    int k = e & 31, f = (e >> 5) & 15, ks = e >> 9;
    wpB[e] = (_Float16)w_proj[(ks * 32 + k) * NF + f];
  }
}

// ---------------------------------------------------------------------------
// Main fused kernel: WG = 4x4x4 voxel tile, 128 threads (4 waves).
// ---------------------------------------------------------------------------
__global__ __launch_bounds__(THREADS)
void sshconv_main(const float* __restrict__ x,
                  const float* __restrict__ w_cp,
                  const float* __restrict__ b_cp,
                  const float* __restrict__ bias,
                  const float* __restrict__ b_proj,
                  const _Float16* __restrict__ ws,
                  float* __restrict__ out) {
  __shared__ __align__(16) unsigned char smem[L_TOTAL];
  _Float16* xt     = (_Float16*)(smem + L_XT);
  _Float16* at     = (_Float16*)(smem + L_AT);
  _Float16* wB_l   = (_Float16*)(smem + L_WB);
  _Float16* wpB_l  = (_Float16*)(smem + L_WPB);
  float*    bias_s = (float*)   (smem + L_BIAS);
  float*    wcp_s  = (float*)   (smem + L_WCP);
  float*    bcp_s  = (float*)   (smem + L_BCP);
  float*    bpj_s  = (float*)   (smem + L_BPJ);
  int*      offtab = (int*)     (smem + L_OFF);
  _Float16* res2   = (_Float16*)(smem + L_RES2);
  _Float16* sbufh  = (_Float16*)(smem + L_SBUF);

  const int tid = threadIdx.x;
  const int blk = blockIdx.x;
  const int bb  = blk / 1728;
  int rem = blk - bb * 1728;
  const int tz = rem / 144; rem -= tz * 144;
  const int ty = rem / 12;
  const int tx = rem - ty * 12;

  // ---- async copy of the f16 constant banks (atoms|wB|wpB) global -> LDS ----
  {
    const unsigned char* gsrc = (const unsigned char*)ws;
    unsigned char* ldst = smem + L_AT;
    for (int i = tid; i < WS_BANK_BYTES / 16; i += THREADS) {
      unsigned loff = (unsigned)(uintptr_t)(ldst + i * 16);   // low32 = LDS offset
      const void* ga = (const void*)(gsrc + i * 16);
      asm volatile("global_load_async_to_lds_b128 %0, %1, off"
                   :: "v"(loff), "v"(ga) : "memory");
    }
  }

  // ---- small f32 params ----
  for (int i = tid; i < 64; i += THREADS) { bias_s[i] = bias[i]; wcp_s[i] = w_cp[i]; }
  if (tid < 16) { bcp_s[tid] = b_cp[tid]; bpj_s[tid] = b_proj[tid]; }

  { // zero res2 (cr=12..15 K-padding for stage-2 A fragments)
    uint4 z = make_uint4(0u, 0u, 0u, 0u);
    uint4* r4 = (uint4*)res2;
    for (int i = tid; i < 4096; i += THREADS) r4[i] = z;
    // zero pad region of xt (halfs 2048..2944): target of padded-tap A loads
    uint4* zp = (uint4*)(xt + 2048);
    for (int i = tid; i < 112; i += THREADS) zp[i] = z;
  }
  { // im2col tap offsets; padded taps point into the zeroed xt pad region
    int hsel = tid >> 6, kse = (tid >> 4) & 3, ie = tid & 15;
    int kl  = hsel * 16 + ie;
    int tap = kse * 32 + kl;
    int az = tap / 25; int ar = tap - az * 25;
    int ay = ar / 5;   int ax = ar - ay * 5;
    offtab[tid] = (tap < 125) ? (((az * 8 + ay) * 8 + ax) * 4) : 2048;
  }
  // 8^3 x 4ch halo tile (SAME padding -> zeros) as f16
  for (int p = tid; p < 512; p += THREADS) {
    int hz = p >> 6, hy = (p >> 3) & 7, hx = p & 7;
    int gz = tz * 4 + hz - 2, gy = ty * 4 + hy - 2, gx = tx * 4 + hx - 2;
    float4 v = make_float4(0.f, 0.f, 0.f, 0.f);
    if ((unsigned)gz < 48u && (unsigned)gy < 48u && (unsigned)gx < 48u) {
      size_t idx = ((((size_t)bb * DIMS + gz) * DIMS + gy) * DIMS + gx) * CIN;
      v = *(const float4*)(x + idx);
    }
    xt[p * 4 + 0] = (_Float16)v.x;
    xt[p * 4 + 1] = (_Float16)v.y;
    xt[p * 4 + 2] = (_Float16)v.z;
    xt[p * 4 + 3] = (_Float16)v.w;
  }
  asm volatile("s_wait_asynccnt 0" ::: "memory");   // async bank copy done
  __syncthreads();

  // ================= Stage 1: WMMA conv against atom bank =================
  const int lane = tid & 31;
  const int wave = tid >> 5;
  const int lh   = lane >> 4;     // lane half
  const int mrow = lane & 15;     // A row / B column within tile
  const int avy = mrow >> 2, avx = mrow & 3;           // wave's z-slice: vz=wave
  const int vbase = ((wave * 8 + avy) * 8 + avx) * 4;

  const v8f vzero = {0.f, 0.f, 0.f, 0.f, 0.f, 0.f, 0.f, 0.f};
  for (int c = 0; c < CIN; c++) {
    v8f acc[6];
#pragma unroll
    for (int nt = 0; nt < 6; nt++) acc[nt] = vzero;

    for (int ks = 0; ks < 4; ks++) {
      v16h afrag;
#pragma unroll
      for (int i = 0; i < 16; i++)
        afrag[i] = xt[vbase + offtab[lh * 64 + ks * 16 + i] + c];
#pragma unroll
      for (int nt = 0; nt < 6; nt++) {
        const v16h bfrag =
            *(const v16h*)(at + ((ks * NATOM + nt * 16 + mrow) * 32 + lh * 16));
        acc[nt] = __builtin_amdgcn_wmma_f32_16x16x32_f16(
            false, afrag, false, bfrag, (short)0, acc[nt], false, false);
      }
    }
    // Scatter C (f16) into [vox][im][kh=col][cr]: no divisions needed.
#pragma unroll
    for (int nt = 0; nt < 6; nt++) {
      const int im = (nt >= 3) ? 1 : 0;
      const int rr = nt - im * 3;
#pragma unroll
      for (int j = 0; j < 8; j++) {
        int vox = wave * 16 + j + lh * 8;
        res2[(((vox * 2 + im) * 16) + mrow) * 16 + (c * 3 + rr)] =
            (_Float16)acc[nt][j];
      }
    }
  }
  __syncthreads();

  // ========== Stage 2: channel mix (WMMA) + spectrum in C-layout ==========
  const int vox2 = wave * 16 + mrow;     // A row voxel for this lane

  // central-pixel term, built directly in C layout as the kh=0 C operand
  v8f cen;
#pragma unroll
  for (int j = 0; j < 8; j++) {
    int vl = j + lh * 8;
    int cb = (((wave + 2) * 8 + ((vl >> 2) + 2)) * 8 + ((vl & 3) + 2)) * 4;
    float a = bcp_s[mrow];
#pragma unroll
    for (int c = 0; c < 4; c++) a += (float)xt[cb + c] * wcp_s[c * NF + mrow];
    cen[j] = a;
  }

  v8f spec[4];
#pragma unroll
  for (int d = 0; d < 4; d++) spec[d] = vzero;

#pragma unroll
  for (int kh = 0; kh < 16; kh++) {
    const int deg = (kh == 0) ? 0 : ((kh < 4) ? 1 : ((kh < 9) ? 2 : 3));
    v8h lore = *(const v8h*)(res2 + (((vox2 * 2 + 0) * 16) + kh) * 16 + lh * 8);
    v8h loim = *(const v8h*)(res2 + (((vox2 * 2 + 1) * 16) + kh) * 16 + lh * 8);
    v16h are, aim;
#pragma unroll
    for (int i = 0; i < 8; i++) {
      are[i] = lore[i]; are[i + 8] = (_Float16)0.f;
      aim[i] = loim[i]; aim[i + 8] = (_Float16)0.f;
    }
    const v8h blo = *(const v8h*)(wB_l + ((kh * NF + mrow) * 32 + lh * 8));
    const v8h bhi = *(const v8h*)(wB_l + ((kh * NF + mrow) * 32 + 16 + lh * 8));
    const v16h bfrag = __builtin_shufflevector(blo, bhi, 0, 1, 2, 3, 4, 5, 6, 7, 8, 9, 10, 11, 12, 13, 14, 15);
    v8f re = __builtin_amdgcn_wmma_f32_16x16x32_f16(
        false, are, false, bfrag, (short)0, (kh == 0) ? cen : vzero, false, false);
    v8f im = __builtin_amdgcn_wmma_f32_16x16x32_f16(
        false, aim, false, bfrag, (short)0, vzero, false, false);
#pragma unroll
    for (int j = 0; j < 8; j++) spec[deg][j] += re[j] * re[j] + im[j] * im[j];
  }

  // spectrum -> signed log1p -> f16 s-buffer (C-layout scatter, intra-wave)
  const float INV[4] = {1.f, 1.f / 3.f, 1.f / 5.f, 1.f / 7.f};
#pragma unroll
  for (int d = 0; d < 4; d++) {
    float bi = bias_s[mrow * 4 + d];
#pragma unroll
    for (int j = 0; j < 8; j++) {
      float s = spec[d][j] * INV[d] + bi;
      s = (s >= 0.f) ? log1pf(s) : -log1pf(-s);
      int vox = wave * 16 + j + lh * 8;
      sbufh[vox * 64 + mrow * 4 + d] = (_Float16)s;
    }
  }

  // ================= Projection 64 -> 16 as WMMA + ReLU ===================
  v8f po;
#pragma unroll
  for (int j = 0; j < 8; j++) po[j] = bpj_s[mrow];
#pragma unroll
  for (int ks2 = 0; ks2 < 2; ks2++) {
    v8h lo = *(const v8h*)(sbufh + vox2 * 64 + ks2 * 32 + lh * 16);
    v8h hi = *(const v8h*)(sbufh + vox2 * 64 + ks2 * 32 + lh * 16 + 8);
    v16h a;
#pragma unroll
    for (int i = 0; i < 8; i++) { a[i] = lo[i]; a[i + 8] = hi[i]; }
    const v16h b = *(const v16h*)(wpB_l + ((ks2 * NF + mrow) * 32 + lh * 16));
    po = __builtin_amdgcn_wmma_f32_16x16x32_f16(
        false, a, false, b, (short)0, po, false, false);
  }
  const int gzw = tz * 4 + wave;
#pragma unroll
  for (int j = 0; j < 8; j++) {
    int vl = j + lh * 8;
    int gy = ty * 4 + (vl >> 2), gx = tx * 4 + (vl & 3);
    size_t ob = ((((size_t)bb * DIMS + gzw) * DIMS + gy) * DIMS + gx) * NF + mrow;
    out[ob] = fmaxf(po[j], 0.f);
  }
}

// ---------------------------------------------------------------------------
extern "C" void kernel_launch(void* const* d_in, const int* in_sizes, int n_in,
                              void* d_out, int out_size, void* d_ws, size_t ws_size,
                              hipStream_t stream) {
  (void)in_sizes; (void)n_in; (void)out_size; (void)ws_size;
  const float* x      = (const float*)d_in[0];
  const float* w      = (const float*)d_in[1];
  const float* w_cp   = (const float*)d_in[2];
  const float* b_cp   = (const float*)d_in[3];
  const float* bias   = (const float*)d_in[4];
  const float* w_proj = (const float*)d_in[5];
  const float* b_proj = (const float*)d_in[6];
  _Float16* ws = (_Float16*)d_ws;             // 45,056 B of workspace
  float* out = (float*)d_out;

  sshconv_setup_atoms<<<1, 128, 0, stream>>>(ws);
  sshconv_setup_weights<<<1, 256, 0, stream>>>(w, w_proj, ws);
  // 2 batches * (48/4)^3 tiles = 3456 workgroups, 128 threads each
  sshconv_main<<<3456, 128, 0, stream>>>(x, w_cp, b_cp, bias, b_proj, ws, out);
}
